// SpatiotemporalGAT_13451837571213
// MI455X (gfx1250) — matrix operand order repaired and checked by reading, not compiled, not run on hardware
//
#include <hip/hip_runtime.h>
#include <hip/hip_bf16.h>
#include <math.h>
#include <stdint.h>

// Problem constants (from reference)
#define Bq   8
#define Gq   8
#define Lq   9          // LAG+1
#define Nq   256
#define EMBq 32
#define HIDq 128
#define LN_TOT (Lq*Nq)  // 2304
#define XTOT (Bq*Lq*Nq) // 18432 (D=1)
#define SLOPEq 0.01f

// workspace float offsets
#define WS_MU    0
#define WS_RSTD  1
#define WS_V0    2      // v[0]
#define WS_V1    3      // v[1..32] -> ws[3..34]
#define WS_S     35     // im bias collapse
#define WS_TW    36     // twc[0..31] -> ws[36..67]
#define WS_TB    68
#define EW_OFF   80                 // G*L*N = 18432 floats (s folded in)
#define T_OFF    (80 + Gq*Lq*Nq)    // 18512: t[g,n], G*N = 2048
#define XS_OFF   (T_OFF + Gq*Nq)    // 20560: X_sum[b,g,n], 16384

#define ATTN_WAVES 4                // 128-thread attention blocks
#define ATTN_CHUNK 16               // rows of i per block

typedef __attribute__((ext_vector_type(16))) _Float16 v16h;
typedef __attribute__((ext_vector_type(8)))  float    v8f;

__device__ inline float wave_max(float v) {
#pragma unroll
  for (int o = 16; o > 0; o >>= 1) v = fmaxf(v, __shfl_xor(v, o, 32));
  return v;
}
__device__ inline float wave_sum(float v) {
#pragma unroll
  for (int o = 16; o > 0; o >>= 1) v += __shfl_xor(v, o, 32);
  return v;
}

// ---------------- Kernel 1: batch-norm stats over all of x (scalar since D=1)
__global__ void k_stats(const float* __restrict__ x, float* __restrict__ ws) {
  __shared__ float s1[256], s2[256];
  int tid = threadIdx.x;
  float a = 0.f, b = 0.f;
  for (int i = tid; i < XTOT; i += 256) { float v = x[i]; a += v; b += v * v; }
  s1[tid] = a; s2[tid] = b; __syncthreads();
  for (int o = 128; o > 0; o >>= 1) {
    if (tid < o) { s1[tid] += s1[tid + o]; s2[tid] += s2[tid + o]; }
    __syncthreads();
  }
  if (tid == 0) {
    float mu = s1[0] / (float)XTOT;
    float var = s2[0] / (float)XTOT - mu * mu;
    ws[WS_MU] = mu;
    ws[WS_RSTD] = rsqrtf(var + 1e-5f);
  }
}

// ---------------- Kernel 2: collapse the activation-free MLPs to vectors
__global__ void k_collapse(const float* __restrict__ im_w1, const float* __restrict__ im_b1,
                           const float* __restrict__ im_w2, const float* __restrict__ im_b2,
                           const float* __restrict__ tf_w1, const float* __restrict__ tf_b1,
                           const float* __restrict__ tf_w2, const float* __restrict__ tf_b2,
                           float* __restrict__ ws) {
  int t = threadIdx.x;
  if (t < 33) {                       // v[i] = im_w1[i,:] . im_w2
    float acc = 0.f;
    for (int h = 0; h < HIDq; h++) acc += im_w1[t * HIDq + h] * im_w2[h];
    ws[WS_V0 + t] = acc;
  } else if (t == 33) {               // s = im_b1 . im_w2 + im_b2
    float acc = 0.f;
    for (int h = 0; h < HIDq; h++) acc += im_b1[h] * im_w2[h];
    ws[WS_S] = acc + im_b2[0];
  } else if (t >= 64 && t < 96) {     // twc[e] = (tf_w1[e,:]+tf_w1[e+32,:]) . tf_w2
    int e = t - 64;
    float acc = 0.f;
    for (int h = 0; h < HIDq; h++)
      acc += (tf_w1[e * HIDq + h] + tf_w1[(e + 32) * HIDq + h]) * tf_w2[h];
    ws[WS_TW + e] = acc;
  } else if (t == 96) {
    float acc = 0.f;
    for (int h = 0; h < HIDq; h++) acc += tf_b1[h] * tf_w2[h];
    ws[WS_TB] = acc + tf_b2[0];
  }
}

// ---------------- Kernel 3: ew[g,l,n] = emb . v[1:] + s ; t[g,n] = E_last . twc + tb
__global__ void k_ew(const float* __restrict__ emb, const float* __restrict__ ws,
                     float* __restrict__ ew, float* __restrict__ tarr) {
  int idx = blockIdx.x * 256 + threadIdx.x;
  if (idx < Gq * Lq * Nq) {
    const float* ep = emb + (size_t)idx * EMBq;
    float acc = 0.f;
#pragma unroll
    for (int e = 0; e < EMBq; e++) acc += ep[e] * ws[WS_V1 + e];
    ew[idx] = acc + ws[WS_S];
  } else if (idx < Gq * Lq * Nq + Gq * Nq) {
    int r = idx - Gq * Lq * Nq;
    int g = r / Nq, n = r % Nq;
    const float* ep = emb + ((((size_t)g * Lq) + (Lq - 1)) * Nq + n) * EMBq;
    float acc = 0.f;
#pragma unroll
    for (int e = 0; e < EMBq; e++) acc += ep[e] * ws[WS_TW + e];
    tarr[r] = acc + ws[WS_TB];
  }
}

// ---------------- Kernel 4: double-softmax attention -> X_sum[b,g,i]
// grid.x = B*G*16 ; block = 128 (4 waves); each block handles 16 rows i,
// each wave handles 4 rows. Per row, the 9x256 adjacency plane A[g,:,i,:]
// is staged into a private LDS buffer with coalesced async copies
// (global_load_async_to_lds_b128), overlapped with pass 1 which needs no A.
__global__ void k_attn(const float* __restrict__ x, const float* __restrict__ A,
                       const float* __restrict__ bn_w, const float* __restrict__ bn_b,
                       const float* __restrict__ a_w, const float* __restrict__ a_b,
                       const float* __restrict__ ws, const float* __restrict__ ew,
                       const float* __restrict__ tarr, float* __restrict__ xsum) {
  __shared__ float Wh[LN_TOT];                         // Wh[l*N+j]
  __shared__ float tp[LN_TOT];                         // t_proxy[l*N+i]
  __shared__ __align__(16) float Alds[ATTN_WAVES * LN_TOT];  // per-wave A plane

  const int bg = blockIdx.x / ATTN_CHUNK;
  const int chunk = blockIdx.x % ATTN_CHUNK;
  const int b = bg / Gq, g = bg % Gq;
  const int tid = threadIdx.x;

  const float mu = ws[WS_MU], rstd = ws[WS_RSTD], v0 = ws[WS_V0];
  const float bnw = bn_w[0], bnb = bn_b[0];

  // stage Wh and t_proxy (s already folded into ew)
  for (int idx = tid; idx < LN_TOT; idx += ATTN_WAVES * 32) {
    int l = idx >> 8, j = idx & 255;
    float xn = (x[((size_t)b * Lq + l) * Nq + j] - mu) * rstd * bnw + bnb;
    Wh[idx] = xn * v0 + ew[((size_t)g * Lq + l) * Nq + j];
    tp[idx] = (l < Lq - 1)
                  ? (xn * v0 + ew[((size_t)g * Lq + (Lq - 1)) * Nq + j])
                  : tarr[g * Nq + j];
  }
  __syncthreads();

  const int wave = tid >> 5, lane = tid & 31;
  const float aw0 = a_w[0], aw1 = a_w[1], abv = a_b[0];
  const float* Ag = A + (size_t)g * Lq * Nq * Nq;
  const int ibase = chunk * ATTN_CHUNK;

  float* myA = &Alds[wave * LN_TOT];
  // low 32 bits of the generic pointer are the LDS byte offset
  const uint32_t lds_base = (uint32_t)(uintptr_t)myA;

  for (int i = ibase + wave; i < ibase + ATTN_CHUNK; i += ATTN_WAVES) {
    // ---- async-stage A[g, lidx, i, 0..255] for all lidx into this wave's LDS
    // 9 rows x 1KB, 512B (b128 x 32 lanes) per issue -> 18 async issues
#pragma unroll
    for (int lidx = 0; lidx < Lq; lidx++) {
      const float* rp = Ag + ((size_t)lidx * Nq + i) * Nq;
#pragma unroll
      for (int h = 0; h < 2; h++) {
        const float* gp = rp + h * 128 + lane * 4;
        uint32_t lo = lds_base + (uint32_t)((lidx * Nq + h * 128 + lane * 4) * 4);
        asm volatile("global_load_async_to_lds_b128 %0, %1, off"
                     :: "v"(lo), "v"(gp) : "memory");
      }
    }

    // ---- pass 1: row max of leaky_relu score over k = l*N + j  (no A needed)
    float m1 = -INFINITY;
    for (int k = lane; k < LN_TOT; k += 32) {
      int l = k >> 8;
      float sc = aw0 * Wh[k] + aw1 * tp[(l << 8) + i] + abv;
      sc = sc > 0.f ? sc : SLOPEq * sc;
      m1 = fmaxf(m1, sc);
    }
    m1 = wave_max(m1);

    // adjacency plane must be resident before pass 2
    asm volatile("s_wait_asynccnt 0x0" ::: "memory");

    // ---- pass 2: den1 = sum exp(score-m1); mm = max exp over masked entries
    float den1 = 0.f, mm = -INFINITY;
    for (int k = lane; k < LN_TOT; k += 32) {
      int l = k >> 8;
      float sc = aw0 * Wh[k] + aw1 * tp[(l << 8) + i] + abv;
      sc = sc > 0.f ? sc : SLOPEq * sc;
      float e = expf(sc - m1);
      den1 += e;
      int jr = k / Lq, lr = k - jr * Lq;  // adjacency flatten order (j-major)
      float msk = myA[(Lq - 1 - lr) * Nq + jr];
      if (msk != 0.f) mm = fmaxf(mm, e);
    }
    den1 = wave_sum(den1);
    mm = wave_max(mm);
    float m2 = (mm > -INFINITY) ? (mm / den1) : -1e9f;

    // ---- pass 3: den2 and masked-weighted numerator with Wh_flat (l-major)
    float den2 = 0.f, num = 0.f;
    for (int k = lane; k < LN_TOT; k += 32) {
      int l = k >> 8;
      float sc = aw0 * Wh[k] + aw1 * tp[(l << 8) + i] + abv;
      sc = sc > 0.f ? sc : SLOPEq * sc;
      float attn = expf(sc - m1) / den1;
      int jr = k / Lq, lr = k - jr * Lq;
      float msk = myA[(Lq - 1 - lr) * Nq + jr];
      float val = (msk != 0.f) ? attn : -1e9f;
      float e2 = expf(val - m2);
      den2 += e2;
      if (msk != 0.f) num += e2 * Wh[k];
    }
    den2 = wave_sum(den2);
    num = wave_sum(num);
    if (lane == 0) xsum[(size_t)bg * Nq + i] = num / den2;
  }
}

// ---------------- Kernel 5: out-MLP via WMMA + residual + (degenerate) LayerNorm
// grid.x = (B*G*N)/16 = 1024 ; block = 256 = 8 waves; wave w owns cols [16w,16w+16)
__global__ void k_mlp_wmma(const float* __restrict__ emb, const float* __restrict__ om_w1,
                           const float* __restrict__ om_b1, const float* __restrict__ om_w2,
                           const float* __restrict__ om_b2, const float* __restrict__ x,
                           const float* __restrict__ bn_w, const float* __restrict__ bn_b,
                           const float* __restrict__ ln_g, const float* __restrict__ ln_b,
                           const float* __restrict__ ws, const float* __restrict__ xsum,
                           float* __restrict__ out) {
  __shared__ _Float16 zh[16][64];   // 16 rows of z, K padded 33 -> 64
  __shared__ float rowsum[16];
  const int tid = threadIdx.x;
  const int m0 = blockIdx.x * 16;

  // stage z rows as f16: [X_sum, E_last(32), zero-pad]
  for (int idx = tid; idx < 16 * 64; idx += 256) {
    int r = idx >> 6, kk = idx & 63;
    int rg = m0 + r;
    int g = (rg / Nq) % Gq, n = rg & (Nq - 1);
    float val = 0.f;
    if (kk == 0) val = xsum[rg];
    else if (kk <= 32) val = emb[((((size_t)g * Lq) + (Lq - 1)) * Nq + n) * EMBq + (kk - 1)];
    zh[r][kk] = (_Float16)val;
  }
  if (tid < 16) rowsum[tid] = 0.f;
  __syncthreads();

  const int wave = tid >> 5, lane = tid & 31;
  const int c0 = wave * 16;
  const int col = c0 + (lane & 15);
  const int hi = (lane >= 16) ? 1 : 0;

  v8f acc = {};
#pragma unroll
  for (int ks = 0; ks < 2; ks++) {
    const int kb = ks * 32;
    v16h a, bf;
    // A frag: 16x32 f16 layout — M = lane%16; K = h + hi*16
#pragma unroll
    for (int h = 0; h < 16; h++) {
      int K = h + (hi << 4);
      a[h] = zh[lane & 15][kb + K];
    }
    // B frag: 32x16 f16 — lanes 0-15: K=h, lanes 16-31: K=h+16; N = col
#pragma unroll
    for (int h = 0; h < 16; h++) {
      int K = kb + h + (hi << 4);
      float w = (K < 33) ? om_w1[K * HIDq + col] : 0.f;
      bf[h] = (_Float16)w;
    }
    acc = __builtin_amdgcn_wmma_f32_16x16x32_f16(false, a, false, bf,
                                                 (short)0, acc, false, false);
  }

  // epilogue: hidden = lrelu(acc + b1); contract with om_w2 column weight
  const float b1 = om_b1[col], w2 = om_w2[col];
#pragma unroll
  for (int r = 0; r < 8; r++) {
    int M = r + (hi << 3);
    float hv = acc[r] + b1;
    hv = hv > 0.f ? hv : SLOPEq * hv;
    atomicAdd(&rowsum[M], hv * w2);   // ds_add_f32
  }
  __syncthreads();

  if (tid < 16) {
    int rg = m0 + tid;
    int b = rg / (Gq * Nq), n = rg & (Nq - 1);
    float mu = ws[WS_MU], rstd = ws[WS_RSTD];
    float residual = (x[((size_t)b * Lq + (Lq - 2)) * Nq + n] - mu) * rstd * bn_w[0] + bn_b[0];
    float y = rowsum[tid] + om_b2[0] + residual;
    // LayerNorm over last dim (D=1): mean==y, var==0 -> result is ln_b exactly
    float mu2 = y;
    float v2 = 0.f;
    float lnv = (y - mu2) * rsqrtf(v2 + 1e-5f) * ln_g[0] + ln_b[0];
    out[rg] = lnv;
  }
}

extern "C" void kernel_launch(void* const* d_in, const int* in_sizes, int n_in,
                              void* d_out, int out_size, void* d_ws, size_t ws_size,
                              hipStream_t stream) {
  const float* x     = (const float*)d_in[0];
  const float* A     = (const float*)d_in[1];
  const float* bn_w  = (const float*)d_in[2];
  const float* bn_b  = (const float*)d_in[3];
  const float* emb   = (const float*)d_in[4];
  const float* im_w1 = (const float*)d_in[5];
  const float* im_b1 = (const float*)d_in[6];
  const float* im_w2 = (const float*)d_in[7];
  const float* im_b2 = (const float*)d_in[8];
  const float* tf_w1 = (const float*)d_in[9];
  const float* tf_b1 = (const float*)d_in[10];
  const float* tf_w2 = (const float*)d_in[11];
  const float* tf_b2 = (const float*)d_in[12];
  const float* om_w1 = (const float*)d_in[13];
  const float* om_b1 = (const float*)d_in[14];
  const float* om_w2 = (const float*)d_in[15];
  const float* om_b2 = (const float*)d_in[16];
  const float* a_w   = (const float*)d_in[17];
  const float* a_b   = (const float*)d_in[18];
  const float* ln_g  = (const float*)d_in[19];
  const float* ln_b  = (const float*)d_in[20];

  float* ws   = (float*)d_ws;
  float* ew   = ws + EW_OFF;
  float* tarr = ws + T_OFF;
  float* xsum = ws + XS_OFF;
  float* out  = (float*)d_out;

  k_stats<<<1, 256, 0, stream>>>(x, ws);
  k_collapse<<<1, 128, 0, stream>>>(im_w1, im_b1, im_w2, im_b2,
                                    tf_w1, tf_b1, tf_w2, tf_b2, ws);
  k_ew<<<(Gq * Lq * Nq + Gq * Nq) / 256, 256, 0, stream>>>(emb, ws, ew, tarr);
  k_attn<<<Bq * Gq * ATTN_CHUNK, ATTN_WAVES * 32, 0, stream>>>(
      x, A, bn_w, bn_b, a_w, a_b, ws, ew, tarr, xsum);
  k_mlp_wmma<<<(Bq * Gq * Nq) / 16, 256, 0, stream>>>(emb, om_w1, om_b1, om_w2, om_b2,
                                                      x, bn_w, bn_b, ln_g, ln_b,
                                                      ws, xsum, out);
}
